// SNN_71786083386209
// MI455X (gfx1250) — matrix operand order repaired and checked by reading, not compiled, not run on hardware
//
#include <hip/hip_runtime.h>

typedef __attribute__((ext_vector_type(16))) _Float16 v16h;
typedef __attribute__((ext_vector_type(8)))  _Float16 v8h;
typedef __attribute__((ext_vector_type(8)))  float    v8f;

// Problem constants (from reference)
#define BATCH 256
#define TSTEPS 100
#define DIN 784
#define KP1 800          // DIN padded to multiple of 32
#define H1 1024
#define H2 512
#define NCLS 10
#define NC3 16           // NCLS padded to 16
#define MROWS (BATCH * TSTEPS)   // 25600

// ---------------------------------------------------------------------------
// Prep: f32 input [B, IN, T] -> f16 A0 [B*T, KP1] (zero K-pad)
// ---------------------------------------------------------------------------
__global__ void snn_cast_input(const float* __restrict__ in, _Float16* __restrict__ a0) {
    long long e = (long long)blockIdx.x * blockDim.x + threadIdx.x;
    const long long total = (long long)MROWS * KP1;
    if (e >= total) return;
    int row = (int)(e / KP1);
    int k   = (int)(e - (long long)row * KP1);
    int b = row / TSTEPS, t = row - b * TSTEPS;
    float v = (k < DIN) ? in[((long long)b * DIN + k) * TSTEPS + t] : 0.0f;
    a0[e] = (_Float16)v;
}

// Prep: f32 W [K, N] -> f16 Wt [NR, Kpad]  (Wt[n*Kpad+k] = W[k*N+n], zero padded)
__global__ void snn_wtrans(const float* __restrict__ w, _Float16* __restrict__ wt,
                           int K, int N, int NR, int Kpad) {
    long long e = (long long)blockIdx.x * blockDim.x + threadIdx.x;
    const long long total = (long long)NR * Kpad;
    if (e >= total) return;
    int n = (int)(e / Kpad);
    int k = (int)(e - (long long)n * Kpad);
    float v = (k < K && n < N) ? w[(long long)k * N + n] : 0.0f;
    wt[e] = (_Float16)v;
}

// ---------------------------------------------------------------------------
// WMMA f16 GEMM:  C[M,N] f32 = A[M,K] f16 (row-major) x Wt[N,K] f16 (W^T rows)
// ASYNC=true : triple-buffered LDS fed by GLOBAL_LOAD_ASYNC_TO_LDS_B128
//              (ASYNCcnt-managed, 1 barrier per k-step, zero staging VGPRs)
// ASYNC=false: classic double-buffered reg-staged path (tiny tail GEMM only)
// ---------------------------------------------------------------------------
template<int BM, int BN, int WAVES_M, int WAVES_N, int MT, int NT, bool ASYNC>
__global__ __launch_bounds__(WAVES_M * WAVES_N * 32)
__attribute__((amdgpu_waves_per_eu(1)))
void snn_gemm_f16(const _Float16* __restrict__ A, const _Float16* __restrict__ Bt,
                  float* __restrict__ C, int M, int N, int K) {
    constexpr int THREADS = WAVES_M * WAVES_N * 32;
    constexpr int LDA = 40;   // halfs per LDS row slot (32 data + 8 pad -> bank spread)
    constexpr int NBUF = ASYNC ? 3 : 2;
    __shared__ __align__(64) _Float16 As[NBUF][BM * LDA];
    __shared__ __align__(64) _Float16 Bs[NBUF][BN * LDA];

    const int tid  = threadIdx.x;
    const int lane = tid & 31;
    const int wave = tid >> 5;
    const int wm = wave % WAVES_M;
    const int wn = wave / WAVES_M;
    const int m0 = blockIdx.y * BM;
    const int n0 = blockIdx.x * BN;

    constexpr int A_EL = BM * 4;   // 16B chunks per A tile
    constexpr int B_EL = BN * 4;
    constexpr int AREG = (A_EL + THREADS - 1) / THREADS;
    constexpr int BREG = (B_EL + THREADS - 1) / THREADS;
    // Wave-uniform ASYNCcnt bookkeeping requires even chunk distribution:
    static_assert(!ASYNC || ((A_EL % THREADS) == 0 && (B_EL % THREADS) == 0), "");
    constexpr int WAITN = AREG + BREG;   // async ops per wave per tile

    const int KT = K >> 5;

    const unsigned long long abase = (unsigned long long)(size_t)A;
    const unsigned long long bbase = (unsigned long long)(size_t)Bt;

    // ---- async tile prefetch: global -> LDS DMA, no VGPR data ----
    auto issue_async = [&](int kt, int bufi) {
        const int k0 = kt << 5;
#pragma unroll
        for (int i = 0; i < AREG; ++i) {
            int e = tid + i * THREADS;
            int r = e >> 2, q = e & 3;
            unsigned goff = (unsigned)(((size_t)(m0 + r) * K + k0 + q * 8) * sizeof(_Float16));
            unsigned ldsa = (unsigned)(size_t)&As[bufi][r * LDA + q * 8];
            asm volatile("global_load_async_to_lds_b128 %0, %1, %2"
                         :: "v"(ldsa), "v"(goff), "s"(abase) : "memory");
        }
#pragma unroll
        for (int i = 0; i < BREG; ++i) {
            int e = tid + i * THREADS;
            int r = e >> 2, q = e & 3;
            unsigned goff = (unsigned)(((size_t)(n0 + r) * K + k0 + q * 8) * sizeof(_Float16));
            unsigned ldsa = (unsigned)(size_t)&Bs[bufi][r * LDA + q * 8];
            asm volatile("global_load_async_to_lds_b128 %0, %1, %2"
                         :: "v"(ldsa), "v"(goff), "s"(bbase) : "memory");
        }
    };

    // ---- sync-path staging (tail GEMM) ----
    uint4 areg[AREG], breg[BREG];
    auto fetch = [&](int kt) {
        const int k0 = kt << 5;
#pragma unroll
        for (int i = 0; i < AREG; ++i) {
            int e = tid + i * THREADS;
            if ((A_EL % THREADS) == 0 || e < A_EL) {
                int r = e >> 2, q = e & 3;
                areg[i] = *(const uint4*)(A + (size_t)(m0 + r) * K + k0 + q * 8);
            }
        }
#pragma unroll
        for (int i = 0; i < BREG; ++i) {
            int e = tid + i * THREADS;
            if ((B_EL % THREADS) == 0 || e < B_EL) {
                int r = e >> 2, q = e & 3;
                breg[i] = *(const uint4*)(Bt + (size_t)(n0 + r) * K + k0 + q * 8);
            }
        }
    };
    auto commit = [&](int bufi) {
#pragma unroll
        for (int i = 0; i < AREG; ++i) {
            int e = tid + i * THREADS;
            if ((A_EL % THREADS) == 0 || e < A_EL) {
                int r = e >> 2, q = e & 3;
                *(uint4*)(&As[bufi][r * LDA + q * 8]) = areg[i];
            }
        }
#pragma unroll
        for (int i = 0; i < BREG; ++i) {
            int e = tid + i * THREADS;
            if ((B_EL % THREADS) == 0 || e < B_EL) {
                int r = e >> 2, q = e & 3;
                *(uint4*)(&Bs[bufi][r * LDA + q * 8]) = breg[i];
            }
        }
    };

    v8f acc[MT][NT] = {};

    const int l    = lane & 15;
    const int uphf = lane >> 4;            // 0: lanes 0-15, 1: lanes 16-31
    const int ka0  = uphf ? 16 : 0;        // A frag: K offsets, contiguous k = 16*uphf + s
    const int ka1  = uphf ? 24 : 8;
    const int kb   = uphf ? 16 : 0;        // B frag: K offset, contiguous k = 16*uphf + s (same as A)

    auto compute = [&](int bufi) {
        v16h afrag[MT];
        v16h bfrag[NT];
#pragma unroll
        for (int mi = 0; mi < MT; ++mi) {
            const _Float16* p = &As[bufi][(wm * MT * 16 + mi * 16 + l) * LDA];
            v8h lo = *(const v8h*)(p + ka0);
            v8h hi = *(const v8h*)(p + ka1);
            afrag[mi] = __builtin_shufflevector(lo, hi, 0,1,2,3,4,5,6,7,8,9,10,11,12,13,14,15);
        }
#pragma unroll
        for (int ni = 0; ni < NT; ++ni) {
            const _Float16* p = &Bs[bufi][(wn * NT * 16 + ni * 16 + l) * LDA + kb];
            v8h lo = *(const v8h*)(p);
            v8h hi = *(const v8h*)(p + 8);
            bfrag[ni] = __builtin_shufflevector(lo, hi, 0,1,2,3,4,5,6,7,8,9,10,11,12,13,14,15);
        }
#pragma unroll
        for (int mi = 0; mi < MT; ++mi)
#pragma unroll
            for (int ni = 0; ni < NT; ++ni)
                acc[mi][ni] = __builtin_amdgcn_wmma_f32_16x16x32_f16(
                    false, afrag[mi], false, bfrag[ni], (short)0, acc[mi][ni], false, false);
    };

    if constexpr (ASYNC) {
        issue_async(0, 0);
        for (int kt = 0; kt < KT; ++kt) {
            const int bufi = kt % 3;
            if (kt + 1 < KT) {
                issue_async(kt + 1, (kt + 1) % 3);
                // tile kt's WAITN ops done once <= WAITN remain (in-order per wave)
                asm volatile("s_wait_asynccnt %0" :: "n"(WAITN) : "memory");
            } else {
                asm volatile("s_wait_asynccnt 0" ::: "memory");
            }
            __syncthreads();   // publish tile kt to all waves; also fences buf reuse
            compute(bufi);
        }
    } else {
        fetch(0);
        commit(0);
        for (int kt = 0; kt < KT; ++kt) {
            const int bufi = kt & 1;
            __syncthreads();
            if (kt + 1 < KT) fetch(kt + 1);
            compute(bufi);
            __syncthreads();
            if (kt + 1 < KT) commit(bufi ^ 1);
        }
    }

    // C/D layout: VGPR r -> row (r + 8*uphf), lane&15 -> col
#pragma unroll
    for (int mi = 0; mi < MT; ++mi) {
        const int rbase = m0 + wm * MT * 16 + mi * 16 + uphf * 8;
#pragma unroll
        for (int ni = 0; ni < NT; ++ni) {
            const int col = n0 + wn * NT * 16 + ni * 16 + l;
            float* cp = C + (size_t)rbase * N + col;
#pragma unroll
            for (int r = 0; r < 8; ++r) cp[(size_t)r * N] = acc[mi][ni][r];
        }
    }
}

// ---------------------------------------------------------------------------
// LIF scan: one thread per neuron (b,d); cur [b*T+t, D] f32 -> spikes f16 same layout
// ---------------------------------------------------------------------------
__global__ void snn_lif(const float* __restrict__ cur, _Float16* __restrict__ spk,
                        int D, int total) {
    int id = blockIdx.x * blockDim.x + threadIdx.x;
    if (id >= total) return;
    int b = id / D, d = id - b * D;
    const float* c = cur + (size_t)b * TSTEPS * D + d;
    _Float16*    s = spk + (size_t)b * TSTEPS * D + d;
    float v = 0.0f, i = 0.0f;
#pragma unroll 4
    for (int t = 0; t < TSTEPS; ++t) {
        float x = c[(size_t)t * D];
        v = 0.95f * v;
        i = 0.8f * i + x;
        v = v + i;
        bool fire = v > 1.0f;
        s[(size_t)t * D] = (_Float16)(fire ? 1.0f : 0.0f);
        v = fire ? 0.0f : v;
    }
}

// Last layer: LIF + sum over time fused. cur [b*T+t, 16] f32 -> out [b, 10] f32
__global__ void snn_lif_sum(const float* __restrict__ cur, float* __restrict__ out) {
    int id = blockIdx.x * blockDim.x + threadIdx.x;
    if (id >= BATCH * NCLS) return;
    int b = id / NCLS, c = id - b * NCLS;
    const float* p = cur + (size_t)b * TSTEPS * NC3 + c;
    float v = 0.0f, i = 0.0f, cnt = 0.0f;
#pragma unroll 4
    for (int t = 0; t < TSTEPS; ++t) {
        float x = p[(size_t)t * NC3];
        v = 0.95f * v;
        i = 0.8f * i + x;
        v = v + i;
        bool fire = v > 1.0f;
        cnt += fire ? 1.0f : 0.0f;
        v = fire ? 0.0f : v;
    }
    out[id] = cnt;
}

// ---------------------------------------------------------------------------
extern "C" void kernel_launch(void* const* d_in, const int* in_sizes, int n_in,
                              void* d_out, int out_size, void* d_ws, size_t ws_size,
                              hipStream_t stream) {
    (void)in_sizes; (void)n_in; (void)out_size; (void)ws_size;
    const float* sp = (const float*)d_in[0];   // [256, 784, 100]
    const float* w1 = (const float*)d_in[1];   // [784, 1024]
    const float* w2 = (const float*)d_in[2];   // [1024, 512]
    const float* w3 = (const float*)d_in[3];   // [512, 10]
    float* out = (float*)d_out;                // [256, 10]

    char* ws = (char*)d_ws;
    size_t off = 0;
    auto alloc = [&](size_t bytes) -> void* {
        void* p = ws + off;
        off = (off + bytes + 255) & ~(size_t)255;
        return p;
    };
    _Float16* A0  = (_Float16*)alloc((size_t)MROWS * KP1 * 2);
    _Float16* A1  = (_Float16*)alloc((size_t)MROWS * H1  * 2);
    _Float16* A2  = (_Float16*)alloc((size_t)MROWS * H2  * 2);
    _Float16* W1t = (_Float16*)alloc((size_t)H1 * KP1 * 2);
    _Float16* W2t = (_Float16*)alloc((size_t)H2 * H1  * 2);
    _Float16* W3t = (_Float16*)alloc((size_t)NC3 * H2 * 2);
    float*    CUR = (float*)   alloc((size_t)MROWS * H1 * 4);   // reused for all layers

    const int TB = 256;

    // --- prep ---
    {
        long long tot = (long long)MROWS * KP1;
        snn_cast_input<<<(unsigned)((tot + TB - 1) / TB), TB, 0, stream>>>(sp, A0);
    }
    snn_wtrans<<<(H1 * KP1 + TB - 1) / TB, TB, 0, stream>>>(w1, W1t, DIN, H1, H1, KP1);
    snn_wtrans<<<(H2 * H1  + TB - 1) / TB, TB, 0, stream>>>(w2, W2t, H1, H2, H2, H1);
    snn_wtrans<<<(NC3 * H2 + TB - 1) / TB, TB, 0, stream>>>(w3, W3t, H2, NCLS, NC3, H2);

    // --- layer 1: GEMM [25600,800]x[800,1024] -> LIF ---
    snn_gemm_f16<128, 128, 4, 2, 2, 4, true>
        <<<dim3(H1 / 128, MROWS / 128), 256, 0, stream>>>(A0, W1t, CUR, MROWS, H1, KP1);
    snn_lif<<<(BATCH * H1 + TB - 1) / TB, TB, 0, stream>>>(CUR, A1, H1, BATCH * H1);

    // --- layer 2: GEMM [25600,1024]x[1024,512] -> LIF ---
    snn_gemm_f16<128, 128, 4, 2, 2, 4, true>
        <<<dim3(H2 / 128, MROWS / 128), 256, 0, stream>>>(A1, W2t, CUR, MROWS, H2, H1);
    snn_lif<<<(BATCH * H2 + TB - 1) / TB, TB, 0, stream>>>(CUR, A2, H2, BATCH * H2);

    // --- layer 3: GEMM [25600,512]x[512,16] -> LIF + time-sum ---
    snn_gemm_f16<256, 16, 8, 1, 2, 1, false>
        <<<dim3(1, MROWS / 256), 256, 0, stream>>>(A2, W3t, CUR, MROWS, NC3, H2);
    snn_lif_sum<<<(BATCH * NCLS + TB - 1) / TB, TB, 0, stream>>>(CUR, out);
}
